// TemporalE8Quantizer_43224550866970
// MI455X (gfx1250) — matrix operand order repaired and checked by reading, not compiled, not run on hardware
//
#include <hip/hip_runtime.h>

// ---------------------------------------------------------------------------
// TemporalE8Quantizer for MI455X (gfx1250, wave32, WMMA).
//
// Roofline: seq (131072x256 f32 = 134MB) dominates; single-pass HBM-bound
// design (~6us floor @ 23.3 TB/s). Big GEMM seq@W1.T done with
// v_wmma_f32_16x16x32_bf16 using an f32 -> bf16 hi/lo split (3 WMMAs:
// AhBh + AhBl + AlBh) for near-f32 precision while staying memory-bound.
// The Wc/Wd chain is algebraically collapsed to one dot with
// v = Wc.T @ Wd[colony_idx] (precomputed on device).
// r_prev / r_curr / ||delta|| fused into the same seq pass.
// ---------------------------------------------------------------------------

#define THRESH     0.7f
#define RISK_JUMP  0.3f
#define SDELTA2    0.25f    // (0.5)^2 : compare squared norm
#define WAVES      8        // 8 waves * 16 rows = 128 output rows / block

typedef __attribute__((ext_vector_type(16))) __bf16          v16bf;
typedef __attribute__((ext_vector_type(16))) unsigned short  v16u;
typedef __attribute__((ext_vector_type(8)))  float           v8f;

static __device__ __forceinline__ unsigned short f2bf_rne(float f) {
    unsigned u = __builtin_bit_cast(unsigned, f);
    u += 0x7FFFu + ((u >> 16) & 1u);       // round-to-nearest-even
    return (unsigned short)(u >> 16);
}
static __device__ __forceinline__ float bf2f(unsigned short h) {
    unsigned u = ((unsigned)h) << 16;
    return __builtin_bit_cast(float, u);
}
static __device__ __forceinline__ float sigmoidf_(float x) {
    return 1.0f / (1.0f + __expf(-x));
}
static __device__ __forceinline__ float gelu_exact(float x) {
    return 0.5f * x * (1.0f + erff(x * 0.70710678118654752f));
}

// ----- E8 lattice rounding (matches reference nearest_e8) -------------------
static __device__ __forceinline__ void nearest_d8(const float* y, float* out) {
    float f[8], err[8];
    float fs = 0.0f;
#pragma unroll
    for (int i = 0; i < 8; ++i) { f[i] = rintf(y[i]); fs += f[i]; }
    float bestA = -1.0f; int idx = 0;
#pragma unroll
    for (int i = 0; i < 8; ++i) {
        err[i] = y[i] - f[i];
        float a = fabsf(err[i]);
        if (a > bestA) { bestA = a; idx = i; }   // first max, like argmax
    }
    bool odd = (fmodf(fs, 2.0f) != 0.0f);
    if (odd) f[idx] += (err[idx] >= 0.0f) ? 1.0f : -1.0f;
#pragma unroll
    for (int i = 0; i < 8; ++i) out[i] = f[i];
}
static __device__ __forceinline__ void nearest_e8(const float* x, float* out) {
    float y[8], a[8], b[8];
#pragma unroll
    for (int i = 0; i < 8; ++i) y[i] = x[i];
    nearest_d8(y, a);
#pragma unroll
    for (int i = 0; i < 8; ++i) y[i] = x[i] - 0.5f;
    nearest_d8(y, b);
    float da = 0.0f, db = 0.0f;
#pragma unroll
    for (int i = 0; i < 8; ++i) {
        b[i] += 0.5f;
        float ea = x[i] - a[i], eb = x[i] - b[i];
        da += ea * ea; db += eb * eb;
    }
#pragma unroll
    for (int i = 0; i < 8; ++i) out[i] = (da <= db) ? a[i] : b[i];
}

// ----- prep: v = Wc.T @ Wd[cidx]; b0; W1 -> bf16 hi/lo ----------------------
__global__ __launch_bounds__(256) void prep_kernel(
    const float* __restrict__ Wc, const float* __restrict__ bc,
    const float* __restrict__ Wd, const float* __restrict__ bd,
    const float* __restrict__ W1, const int* __restrict__ cidxp,
    float* __restrict__ v, float* __restrict__ b0,
    unsigned short* __restrict__ W1h, unsigned short* __restrict__ W1l)
{
    const int tid  = threadIdx.x;       // 0..255 == feature dim d
    const int cidx = *cidxp;
    const float* wdrow = Wd + cidx * 64;
    float acc = 0.0f;
#pragma unroll 8
    for (int c = 0; c < 64; ++c) acc += Wc[c * 256 + tid] * wdrow[c];
    v[tid] = acc;
    if (tid == 0) {
        float s = bd[cidx];
        for (int c = 0; c < 64; ++c) s += bc[c] * wdrow[c];
        *b0 = s;
    }
    // W1 (64x256) f32 -> bf16 hi + bf16 residual
#pragma unroll 4
    for (int t = 0; t < 64; ++t) {
        int idx = t * 256 + tid;
        float x = W1[idx];
        unsigned short hb = f2bf_rne(x);
        W1h[idx] = hb;
        W1l[idx] = f2bf_rne(x - bf2f(hb));
    }
}

// ----- main fused kernel ----------------------------------------------------
__global__ __launch_bounds__(256) void te8q_main_kernel(
    const float* __restrict__ seq,   const float* __restrict__ W2,
    const float* __restrict__ b2,    const float* __restrict__ emb,
    const int*   __restrict__ cidxp, const float* __restrict__ b1,
    const float* __restrict__ vws,   const float* __restrict__ b0ws,
    const unsigned short* __restrict__ W1h,
    const unsigned short* __restrict__ W1l,
    float* __restrict__ out, int S)
{
    __shared__ float vsh[256];
    __shared__ float hbuf[WAVES][16][65];    // pad 65 -> conflict-free row reads

    const int tid  = threadIdx.x;
    const int w    = tid >> 5;
    const int lane = tid & 31;
    const int m    = lane & 15;              // row within 16-row tile
    const int half = lane >> 4;              // K-slice half per WMMA A layout

    vsh[tid] = vws[tid];
    __syncthreads();

    const int nOut = S - 1;
    const int g    = blockIdx.x * (WAVES * 16) + w * 16;   // first output row j0
    int row = g + 1 + m;                     // seq row feeding h / r_curr
    if (row > S - 1) row = S - 1;            // clamp (results guarded below)
    const float* rp0 = seq + (size_t)row * 256;
    const float* pp0 = seq + (size_t)(row - 1) * 256;      // prev row (cache hit)
    const float  b0  = b0ws[0];

    float rdot = 0.0f, pdot = 0.0f, dsq = 0.0f;
    v8f acc[4] = {};                          // 4 N-tiles of 16 -> N=64

#pragma unroll
    for (int kt = 0; kt < 8; ++kt) {          // K = 256 in steps of 32
        const int K0 = kt * 32;
        // A-fragment K slices (contiguous, matching B):
        //   lanes 0-15 : K0+0..15
        //   lanes 16-31: K0+16..31
        const int c0 = K0 + (half ? 16 : 0);
        const int c1 = c0 + 8;

        float a[16], p[16];
        *(float4*)&a[0]  = *(const float4*)(rp0 + c0);
        *(float4*)&a[4]  = *(const float4*)(rp0 + c0 + 4);
        *(float4*)&a[8]  = *(const float4*)(rp0 + c1);
        *(float4*)&a[12] = *(const float4*)(rp0 + c1 + 4);
        *(float4*)&p[0]  = *(const float4*)(pp0 + c0);
        *(float4*)&p[4]  = *(const float4*)(pp0 + c0 + 4);
        *(float4*)&p[8]  = *(const float4*)(pp0 + c1);
        *(float4*)&p[12] = *(const float4*)(pp0 + c1 + 4);

        // fused risk dots + state-delta norm
#pragma unroll
        for (int e = 0; e < 8; ++e) {
            rdot += a[e] * vsh[c0 + e] + a[8 + e] * vsh[c1 + e];
            pdot += p[e] * vsh[c0 + e] + p[8 + e] * vsh[c1 + e];
        }
#pragma unroll
        for (int e = 0; e < 16; ++e) { float d = a[e] - p[e]; dsq += d * d; }

        // f32 -> bf16 hi/lo split of the A fragment
        v16u hb, lb;
#pragma unroll
        for (int e = 0; e < 16; ++e) {
            unsigned short h16 = f2bf_rne(a[e]);
            hb[e] = h16;
            lb[e] = f2bf_rne(a[e] - bf2f(h16));
        }
        v16bf Ah = __builtin_bit_cast(v16bf, hb);
        v16bf Al = __builtin_bit_cast(v16bf, lb);

        // B fragment: lane n=(l&15), 16 contiguous K; lanes>=16 take K+16
        const int kb = K0 + (half ? 16 : 0);
#pragma unroll
        for (int nt = 0; nt < 4; ++nt) {
            const int n = nt * 16 + m;
            v16bf Bh = __builtin_bit_cast(v16bf, *(const v16u*)(W1h + n * 256 + kb));
            v16bf Bl = __builtin_bit_cast(v16bf, *(const v16u*)(W1l + n * 256 + kb));
            // split-precision product: AhBh + AhBl + AlBh
            acc[nt] = __builtin_amdgcn_wmma_f32_16x16x32_bf16(
                false, Al, false, Bh, (short)0, acc[nt], false, false);
            acc[nt] = __builtin_amdgcn_wmma_f32_16x16x32_bf16(
                false, Ah, false, Bl, (short)0, acc[nt], false, false);
            acc[nt] = __builtin_amdgcn_wmma_f32_16x16x32_bf16(
                false, Ah, false, Bh, (short)0, acc[nt], false, false);
        }
    }

    // combine the two K-halves (lane l and l^16 hold same row)
    rdot += __shfl_xor(rdot, 16, 32);
    pdot += __shfl_xor(pdot, 16, 32);
    dsq  += __shfl_xor(dsq, 16, 32);

    const float rc  = sigmoidf_(rdot + b0);   // r_curr for output row j = g+m
    const float rpv = sigmoidf_(pdot + b0);   // r_prev
    const bool crosses = ((rpv < THRESH) && (THRESH <= rc)) ||
                         ((rpv > THRESH) && (THRESH >= rc));
    const bool jump    = fabsf(rc - rpv) > RISK_JUMP;
    const bool mk      = crosses || jump || (dsq > SDELTA2);
    const int  j       = g + m;

    if (half == 0 && j < nOut) {
        out[(size_t)nOut * 8 + j] = rc;                 // r_curr
        out[(size_t)nOut * 9 + j] = mk ? 1.0f : 0.0f;   // mask
    }

    // gelu(h + b1) -> LDS (C layout: vgpr i -> row i+8*half, col nt*16+m)
#pragma unroll
    for (int nt = 0; nt < 4; ++nt) {
        const int n  = nt * 16 + m;
        const float bb = b1[n];
#pragma unroll
        for (int i = 0; i < 8; ++i) {
            float x = acc[nt][i] + bb;
            hbuf[w][i + 8 * half][n] = gelu_exact(x);
        }
    }
    __syncthreads();

    // tail: s8 = h @ W2.T + b2 + emb, scale by (1+r_curr), E8 round, mask
    if (half == 0 && j < nOut) {
        const int cidx = *cidxp;
        const float* hr = hbuf[w][m];
        float s8[8];
#pragma unroll
        for (int o = 0; o < 8; ++o) {
            const float* w2 = W2 + o * 64;
            float s = 0.0f;
#pragma unroll 8
            for (int n = 0; n < 64; ++n) s += hr[n] * w2[n];
            s8[o] = (s + b2[o] + emb[cidx * 8 + o]) * (1.0f + rc);
        }
        float e8[8];
        nearest_e8(s8, e8);
#pragma unroll
        for (int o = 0; o < 8; ++o)
            out[(size_t)j * 8 + o] = mk ? e8[o] : 0.0f;
    }
}

// ---------------------------------------------------------------------------
extern "C" void kernel_launch(void* const* d_in, const int* in_sizes, int n_in,
                              void* d_out, int out_size, void* d_ws, size_t ws_size,
                              hipStream_t stream) {
    const float* seq  = (const float*)d_in[0];   // (1, S, 256)
    const float* Wc   = (const float*)d_in[1];   // (64, 256)
    const float* bc   = (const float*)d_in[2];   // (64,)
    const float* Wd   = (const float*)d_in[3];   // (7, 64)
    const float* bd   = (const float*)d_in[4];   // (7,)
    const float* W1   = (const float*)d_in[5];   // (64, 256)
    const float* b1   = (const float*)d_in[6];   // (64,)
    const float* W2   = (const float*)d_in[7];   // (8, 64)
    const float* b2   = (const float*)d_in[8];   // (8,)
    const float* emb  = (const float*)d_in[9];   // (7, 8)
    const int*   cidx = (const int*)d_in[10];    // scalar

    const int S = in_sizes[0] / 256;
    float* out = (float*)d_out;

    // workspace layout
    char* wsb = (char*)d_ws;
    float*          v   = (float*)(wsb);                 // 256 f32
    float*          b0  = (float*)(wsb + 1024);          // 1 f32
    unsigned short* W1h = (unsigned short*)(wsb + 2048);           // 16384 bf16
    unsigned short* W1l = (unsigned short*)(wsb + 2048 + 32768);   // 16384 bf16

    prep_kernel<<<1, 256, 0, stream>>>(Wc, bc, Wd, bd, W1, cidx, v, b0, W1h, W1l);

    const int nOut   = S - 1;
    const int blocks = (nOut + WAVES * 16 - 1) / (WAVES * 16);
    te8q_main_kernel<<<blocks, 256, 0, stream>>>(
        seq, W2, b2, emb, cidx, b1, v, b0, W1h, W1l, out, S);
}
